// CrossPositionalAttention_28776280883834
// MI455X (gfx1250) — matrix operand order repaired and checked by reading, not compiled, not run on hardware
//
#include <hip/hip_runtime.h>

// ---------------------------------------------------------------------------
// CrossPositionalAttention for MI455X (gfx1250, wave32, WMMA)
//   out = softmax((F WM)(F WN)^T) (F WV) + F      B=8, T=2048, C=64, fp32 io
// Flash attention computed TRANSPOSED: S^T = K Q^T, O^T = V^T P^T.
// Softmax rows live on lanes -> in-lane reductions + one half-swap shuffle;
// P^T feeds the second WMMA as a B-operand directly (no LDS, no barriers).
// ---------------------------------------------------------------------------

typedef __attribute__((ext_vector_type(16))) __bf16 v16bf;
typedef __attribute__((ext_vector_type(8)))  float  v8f;

union AV {            // one WMMA 16-bit operand = 8 VGPRs = 32 bytes/lane
    v16bf    v;
    uint4    u[2];
    unsigned d[8];
};

static constexpr int Bn = 8;
static constexpr int Tn = 2048;
static constexpr int Cn = 64;

__device__ __forceinline__ unsigned f2bf_pair(float a, float b) {
    // round-to-nearest-even f32 -> bf16, packed {lo=a, hi=b}
    union { float f; unsigned u; } x, y; x.f = a; y.f = b;
    unsigned ra = (x.u + 0x7FFFu + ((x.u >> 16) & 1u)) >> 16;
    unsigned rb = (y.u + 0x7FFFu + ((y.u >> 16) & 1u)) >> 16;
    return (ra & 0xFFFFu) | (rb << 16);
}

__device__ __forceinline__ __bf16 f2bf(float f) {
    union { float f; unsigned u; } in; in.f = f;
    unsigned r = (in.u + 0x7FFFu + ((in.u >> 16) & 1u)) >> 16;
    union { unsigned short s; __bf16 b; } out; out.s = (unsigned short)r;
    return out.b;
}

__device__ __forceinline__ v8f wmma_bf16(v16bf a, v16bf b, v8f c) {
    // D = A(16x32 bf16) * B(32x16 bf16) + C(16x16 f32)
    return __builtin_amdgcn_wmma_f32_16x16x32_bf16(
        /*neg_a=*/false, a, /*neg_b=*/false, b,
        /*c_mod=*/(short)0, c, /*reuse_a=*/false, /*reuse_b=*/false);
}

// ---------------------------------------------------------------------------
// Kernel 1: pointwise projections.  Q,K row-major bf16 [B*T,64]; V transposed
// bf16 Vt[b][c][t] so the attention A-operand (V^T tile) loads contiguously.
// ---------------------------------------------------------------------------
__global__ void __launch_bounds__(256)
proj_kernel(const float* __restrict__ F,
            const float* __restrict__ WM,
            const float* __restrict__ WN,
            const float* __restrict__ WV,
            __bf16* __restrict__ Qb,
            __bf16* __restrict__ Kb,
            __bf16* __restrict__ Vt) {
    __shared__ float sW[3 * Cn * Cn];           // 48 KB of the 320 KB LDS
    const int tid = threadIdx.x;
    for (int i = tid; i < Cn * Cn; i += 256) {
        sW[i]               = WM[i];
        sW[Cn * Cn + i]     = WN[i];
        sW[2 * Cn * Cn + i] = WV[i];
    }
    __syncthreads();

    const int col = tid & 63;
    const int row = blockIdx.x * 4 + (tid >> 6);        // global row in [0, B*T)
    const float* __restrict__ frow = F + (size_t)row * Cn;

    float aq = 0.f, ak = 0.f, av = 0.f;
#pragma unroll
    for (int k = 0; k < Cn; ++k) {
        const float f = frow[k];                         // broadcast within row-group
        aq = fmaf(f, sW[k * Cn + col], aq);
        ak = fmaf(f, sW[Cn * Cn + k * Cn + col], ak);
        av = fmaf(f, sW[2 * Cn * Cn + k * Cn + col], av);
    }
    Qb[(size_t)row * Cn + col] = f2bf(aq);
    Kb[(size_t)row * Cn + col] = f2bf(ak);
    const int b = row / Tn, t = row % Tn;
    Vt[((size_t)b * Cn + col) * Tn + t] = f2bf(av);
}

// ---------------------------------------------------------------------------
// Kernel 2: transposed flash attention.  128 threads = 4 independent waves;
// one wave = 16 query rows.  grid = (T/64, B).  KV in chunks of 32.
// ---------------------------------------------------------------------------
__global__ void __launch_bounds__(128)
attn_kernel(const float* __restrict__ F,
            const __bf16* __restrict__ Qb,
            const __bf16* __restrict__ Kb,
            const __bf16* __restrict__ Vt,
            float* __restrict__ out) {
    const int tid  = threadIdx.x;
    const int wave = tid >> 5;
    const int lane = tid & 31;
    const int hi   = lane >> 4;        // half-wave select
    const int lo   = hi ^ 1;
    const int l16  = lane & 15;
    const int b    = blockIdx.y;
    const int q0   = blockIdx.x * 64 + wave * 16;

    // ---- Q^T as B-operand (64ch x 16q), loaded once.
    // ISA 16-bit B-layout: lane = column n (query), K (channel) 0..7|16..23 on lo,
    // half-wave hi gets 8..15|24..31.
    const __bf16* __restrict__ qrow = Qb + ((size_t)b * Tn + q0 + l16) * Cn + hi * 8;
    AV qb0, qb1;
    qb0.u[0] = *(const uint4*)(qrow);      qb0.u[1] = *(const uint4*)(qrow + 16);
    qb1.u[0] = *(const uint4*)(qrow + 32); qb1.u[1] = *(const uint4*)(qrow + 48);

    v8f O0 = {}, O1 = {}, O2 = {}, O3 = {};   // O^T: 64ch x 16q fp32
    float m = -3.0e38f, l = 0.f;              // per-lane (= per-query) state

    const __bf16* __restrict__ kbatch = Kb + (size_t)b * Tn * Cn;
    const __bf16* __restrict__ vbatch = Vt + (size_t)b * Cn * Tn;

    for (int kv0 = 0; kv0 < Tn; kv0 += 32) {
        // ---- A-operands: K rows (16kv x 64ch), two kv tiles.
        // ISA 16-bit A-layout: lane(lo): K=0..7|16..23 ; lane(hi): 8..15|24..31
        const __bf16* krow = kbatch + (size_t)(kv0 + l16) * Cn;
        const __bf16* krw2 = krow + 16 * Cn;
        AV kaA0, kaA1, kaB0, kaB1;
        kaA0.u[0] = *(const uint4*)(krow + hi * 8);
        kaA0.u[1] = *(const uint4*)(krow + 16 + hi * 8);
        kaA1.u[0] = *(const uint4*)(krow + 32 + hi * 8);
        kaA1.u[1] = *(const uint4*)(krow + 48 + hi * 8);
        kaB0.u[0] = *(const uint4*)(krw2 + hi * 8);
        kaB0.u[1] = *(const uint4*)(krw2 + 16 + hi * 8);
        kaB1.u[0] = *(const uint4*)(krw2 + 32 + hi * 8);
        kaB1.u[1] = *(const uint4*)(krw2 + 48 + hi * 8);

        if (kv0 + 32 < Tn) {                       // prefetch next chunk
            __builtin_prefetch(krow + 32 * Cn, 0, 1);
            __builtin_prefetch(vbatch + (size_t)l16 * Tn + kv0 + 32, 0, 1);
        }

        v8f sA = {}, sB = {};                      // S^T tiles: kv x q
        sA = wmma_bf16(kaA0.v, qb0.v, sA);         // channels  0..31
        sA = wmma_bf16(kaA1.v, qb1.v, sA);         // channels 32..63
        sB = wmma_bf16(kaB0.v, qb0.v, sB);
        sB = wmma_bf16(kaB1.v, qb1.v, sB);

        // ---- online softmax: row (fixed q) = fixed lane; kv across VGPRs
        // (this half: sA->kv 0..7 / 8..15, sB->kv 16..23 / 24..31) + partner.
        float tm = fmaxf(sA[0], sB[0]);
#pragma unroll
        for (int r = 1; r < 8; ++r) tm = fmaxf(tm, fmaxf(sA[r], sB[r]));
        tm = fmaxf(tm, __shfl_xor(tm, 16, 32));    // combine half-waves
        const float nm   = fmaxf(m, tm);
        const float corr = __expf(m - nm);
        m = nm;

        float eA[8], eB[8], rs = 0.f;
#pragma unroll
        for (int r = 0; r < 8; ++r) {
            eA[r] = __expf(sA[r] - nm);
            eB[r] = __expf(sB[r] - nm);
            rs += eA[r] + eB[r];
        }
        rs += __shfl_xor(rs, 16, 32);
        l = l * corr + rs;

#pragma unroll
        for (int r = 0; r < 8; ++r) { O0[r] *= corr; O1[r] *= corr; O2[r] *= corr; O3[r] *= corr; }

        // ---- build P^T B-operand (32kv x 16q) in-register:
        // lo lanes need kv 0..15 (own eA + partner eA), hi lanes kv 16..31.
        AV pb;
#pragma unroll
        for (int j = 0; j < 4; ++j) {
            const unsigned pa  = f2bf_pair(eA[2 * j], eA[2 * j + 1]);
            const unsigned pk  = f2bf_pair(eB[2 * j], eB[2 * j + 1]);
            const unsigned x   = lo ? pk : pa;                    // what partner needs
            const unsigned xs  = (unsigned)__shfl_xor((int)x, 16, 32);
            pb.d[j]     = lo ? pa : xs;    // kv {2j,2j+1}   | {16+2j,16+2j+1}
            pb.d[4 + j] = lo ? xs : pk;    // kv {8+2j,...}  | {24+2j,...}
        }

        // ---- O^T += V^T(ch-tile x 32kv) * P^T, 4 channel tiles
        const __bf16* vp = vbatch + (size_t)l16 * Tn + kv0 + hi * 16;
        AV va0, va1, va2, va3;
        va0.u[0] = *(const uint4*)(vp);
        va0.u[1] = *(const uint4*)(vp + 8);
        va1.u[0] = *(const uint4*)(vp + 16 * Tn);
        va1.u[1] = *(const uint4*)(vp + 16 * Tn + 8);
        va2.u[0] = *(const uint4*)(vp + 32 * Tn);
        va2.u[1] = *(const uint4*)(vp + 32 * Tn + 8);
        va3.u[0] = *(const uint4*)(vp + 48 * Tn);
        va3.u[1] = *(const uint4*)(vp + 48 * Tn + 8);

        O0 = wmma_bf16(va0.v, pb.v, O0);
        O1 = wmma_bf16(va1.v, pb.v, O1);
        O2 = wmma_bf16(va2.v, pb.v, O2);
        O3 = wmma_bf16(va3.v, pb.v, O3);
    }

    // ---- epilogue: O^T D-layout -> out[q][ch], normalize + residual (fp32)
    const float inv = 1.0f / l;
    const size_t rowbase = ((size_t)b * Tn + q0 + l16) * Cn;
#pragma unroll
    for (int r = 0; r < 8; ++r) {
        const int ch = r + hi * 8;
        out[rowbase + ch]      = O0[r] * inv + F[rowbase + ch];
        out[rowbase + ch + 16] = O1[r] * inv + F[rowbase + ch + 16];
        out[rowbase + ch + 32] = O2[r] * inv + F[rowbase + ch + 32];
        out[rowbase + ch + 48] = O3[r] * inv + F[rowbase + ch + 48];
    }
}

// ---------------------------------------------------------------------------
extern "C" void kernel_launch(void* const* d_in, const int* in_sizes, int n_in,
                              void* d_out, int out_size, void* d_ws, size_t ws_size,
                              hipStream_t stream) {
    (void)in_sizes; (void)n_in; (void)out_size; (void)ws_size;
    const float* F  = (const float*)d_in[0];
    const float* WM = (const float*)d_in[1];
    const float* WN = (const float*)d_in[2];
    const float* WV = (const float*)d_in[3];
    float* out = (float*)d_out;

    const size_t elems = (size_t)Bn * Tn * Cn;          // 1M
    __bf16* Qb = (__bf16*)d_ws;                          // 2 MB
    __bf16* Kb = Qb + elems;                             // 2 MB
    __bf16* Vt = Kb + elems;                             // 2 MB (transposed V)

    proj_kernel<<<dim3(Bn * Tn / 4), 256, 0, stream>>>(F, WM, WN, WV, Qb, Kb, Vt);
    attn_kernel<<<dim3(Tn / 64, Bn), 128, 0, stream>>>(F, Qb, Kb, Vt, out);
}
